// WindowAttention_28226525070366
// MI455X (gfx1250) — matrix operand order repaired and checked by reading, not compiled, not run on hardware
//
#include <hip/hip_runtime.h>
#include <hip/hip_bf16.h>

// ---------------------------------------------------------------------------
// Swin window attention, fused, CDNA5 (gfx1250) WMMA f16/f32.
// B=4096 windows, N=49 tokens (padded to 64), C=384, H=12, Dh=32.
// One 256-thread block (8 wave32) per window; 1x4 N-tile blocking in the
// dense GEMMs so each A-fragment LDS read feeds 4 v_wmma ops.
// ---------------------------------------------------------------------------

#define DIMC   384
#define NTOK   49
#define NHEAD  12
#define DH     32
#define QKVC   1152

// padded LDS strides (f16 elements); both are multiples of 8 (16B aligned rows)
#define XS_STRIDE 392   // 384 + 8 pad
#define VT_STRIDE 72    // 64 + 8 pad

typedef __attribute__((ext_vector_type(16))) _Float16 v16h;
typedef __attribute__((ext_vector_type(8)))  _Float16 v8h;
typedef __attribute__((ext_vector_type(8)))  float    v8f;

__device__ __forceinline__ v16h cat16(v8h lo, v8h hi) {
  return __builtin_shufflevector(lo, hi, 0,1,2,3,4,5,6,7,8,9,10,11,12,13,14,15);
}

// A fragment: source row-major [16 x K], pointer already at (row0, kbase).
// Lane half h (lane>>4), m = lane&15 holds K = [8h,8h+8) and [16+8h,16+8h+8).
__device__ __forceinline__ v16h load_fragA(const _Float16* src, int stride, int lane) {
  const int half = lane >> 4, m = lane & 15;
  const _Float16* p = src + m * stride + 8 * half;
  v8h lo = *(const v8h*)(p);
  v8h hi = *(const v8h*)(p + 16);
  return cat16(lo, hi);
}

// B fragment: source is B^T row-major [16(N) x K], pointer at (n0, kbase).
// Lane half h, n = lane&15 holds K = [8h,8h+8) and [16+8h,16+8h+8).
__device__ __forceinline__ v16h load_fragB(const _Float16* srcT, int stride, int lane) {
  const int half = lane >> 4, n = lane & 15;
  const _Float16* p = srcT + n * stride + 8 * half;
  v8h lo = *(const v8h*)(p);
  v8h hi = *(const v8h*)(p + 16);
  return cat16(lo, hi);
}

__device__ __forceinline__ v8f wmma_f16(v16h a, v16h b, v8f c) {
  return __builtin_amdgcn_wmma_f32_16x16x32_f16(false, a, false, b, (short)0, c,
                                                false, false);
}

// ---------------------------------------------------------------------------
// Prep: weight transpose+f16 convert, bias materialization (one-time, tiny).
// ---------------------------------------------------------------------------
__global__ void swin_prep_kernel(const float* __restrict__ qkv_w,
                                 const float* __restrict__ proj_w,
                                 const float* __restrict__ bias_table,
                                 const int*   __restrict__ rel_index,
                                 _Float16* __restrict__ wT,
                                 _Float16* __restrict__ pT,
                                 float* __restrict__ biasbuf) {
  const int NW = QKVC * DIMC;          // 442368
  const int NP = DIMC * DIMC;          // 147456
  const int NB = NHEAD * 64 * 64;      // 49152
  int i = blockIdx.x * 256 + threadIdx.x;
  if (i < NW) {
    int c = i / DIMC, k = i % DIMC;
    wT[i] = (_Float16)qkv_w[k * QKVC + c];
  } else if (i < NW + NP) {
    int j = i - NW;
    int c = j / DIMC, k = j % DIMC;
    pT[j] = (_Float16)proj_w[k * DIMC + c];
  } else if (i < NW + NP + NB) {
    int j = i - NW - NP;
    int h = j >> 12, m = (j >> 6) & 63, n = j & 63;
    float v;
    if (n >= NTOK)      v = -1e30f;                     // mask padded keys
    else if (m >= NTOK) v = 0.f;                        // padded query rows: don't care
    else                v = bias_table[rel_index[m * NTOK + n] * NHEAD + h];
    biasbuf[j] = v;
  }
}

// ---------------------------------------------------------------------------
// Fused window attention: one block (8 wave32) per window.
// ---------------------------------------------------------------------------
__global__ __launch_bounds__(256, 1)
void swin_attn_kernel(const float* __restrict__ x,
                      const float* __restrict__ qkv_b,
                      const float* __restrict__ proj_b,
                      const _Float16* __restrict__ wT,   // [1152][384] f16 (= qkv_w^T)
                      const _Float16* __restrict__ pT,   // [384][384]  f16 (= proj_w^T)
                      const float* __restrict__ biasbuf, // [12][64][64] f32
                      float* __restrict__ out) {
  extern __shared__ _Float16 smem[];
  _Float16* xs = smem;                    // [64][XS_STRIDE]  x, later attn-out
  _Float16* qs = xs + 64 * XS_STRIDE;     // [64][XS_STRIDE]  q (pre-scaled)
  _Float16* ks = qs + 64 * XS_STRIDE;     // [64][XS_STRIDE]  k
  _Float16* vT = ks + 64 * XS_STRIDE;     // [384][VT_STRIDE] v transposed
  _Float16* pb = vT + 384 * VT_STRIDE;    // [8][16][VT_STRIDE] per-wave probs

  const int b    = blockIdx.x;
  const int tid  = threadIdx.x;
  const int lane = tid & 31;
  const int wave = tid >> 5;
  const int half = lane >> 4;
  const int nlan = lane & 15;
  const float scale = 0.17677669529663689f;   // 32^-0.5

  // ---- Stage x[b] -> LDS f16, zero padded rows -----------------------------
  const float* xb = x + (size_t)b * NTOK * DIMC;
  for (int i = tid; i < 64 * (DIMC / 4); i += 256) {
    int m = i / (DIMC / 4), k4 = (i % (DIMC / 4)) * 4;
    float4 v = (m < NTOK) ? *(const float4*)(xb + m * DIMC + k4)
                          : make_float4(0.f, 0.f, 0.f, 0.f);
    _Float16* d = xs + m * XS_STRIDE + k4;
    d[0] = (_Float16)v.x; d[1] = (_Float16)v.y;
    d[2] = (_Float16)v.z; d[3] = (_Float16)v.w;
  }
  __syncthreads();

  // ---- QKV GEMM: [64x384] @ [384x1152] -------------------------------------
  // 4 M-tiles x 72 N-tiles, grouped 4 N-tiles per wave-iteration so each
  // A-fragment feeds 4 WMMAs. 72 groups over 8 waves (uniform q/k/v per group).
  for (int tg = wave; tg < 4 * 18; tg += 8) {
    const int mt  = tg / 18;
    const int nt0 = (tg % 18) * 4;
    v8f acc[4] = {{}, {}, {}, {}};
    for (int kt = 0; kt < DIMC / 32; ++kt) {
      v16h a = load_fragA(xs + mt * 16 * XS_STRIDE + kt * 32, XS_STRIDE, lane);
#pragma unroll
      for (int j = 0; j < 4; ++j) {
        v16h bf = load_fragB(wT + (nt0 + j) * 16 * DIMC + kt * 32, DIMC, lane);
        acc[j] = wmma_f16(a, bf, acc[j]);
      }
    }
#pragma unroll
    for (int j = 0; j < 4; ++j) {
      const int col   = (nt0 + j) * 16 + nlan;   // 0..1151
      const float bv  = qkv_b[col];
      const int which = col / DIMC;              // 0=q 1=k 2=v (uniform per tile)
      const int rem   = col % DIMC;              // h*32 + d
#pragma unroll
      for (int r = 0; r < 8; ++r) {
        const int m = mt * 16 + r + 8 * half;
        const float v = acc[j][r] + bv;
        if (which == 0)      qs[m * XS_STRIDE + rem] = (_Float16)(v * scale);
        else if (which == 1) ks[m * XS_STRIDE + rem] = (_Float16)v;
        else                 vT[rem * VT_STRIDE + m] = (_Float16)v;
      }
    }
  }
  __syncthreads();

  // ---- Attention: head-per-wave, S = qk^T + bias, softmax, O = P V ---------
  _Float16* pw = pb + wave * 16 * VT_STRIDE;
  for (int h = wave; h < NHEAD; h += 8) {
    const float* bh = biasbuf + h * 64 * 64;
    for (int mt = 0; mt < 4; ++mt) {
      v16h aq = load_fragA(qs + mt * 16 * XS_STRIDE + h * DH, XS_STRIDE, lane);
      v8f s[4];
#pragma unroll
      for (int nt = 0; nt < 4; ++nt) {
        v16h bk = load_fragB(ks + nt * 16 * XS_STRIDE + h * DH, XS_STRIDE, lane);
        v8f z = {};
        s[nt] = wmma_f16(aq, bk, z);
      }
      // + relative position bias (padded cols carry -1e30)
#pragma unroll
      for (int nt = 0; nt < 4; ++nt)
#pragma unroll
        for (int r = 0; r < 8; ++r) {
          const int m = mt * 16 + r + 8 * half;
          s[nt][r] += bh[m * 64 + nt * 16 + nlan];
        }
      // row softmax: reduce over 4 reg tiles + 16 lanes of the half-wave
#pragma unroll
      for (int r = 0; r < 8; ++r) {
        float mx = fmaxf(fmaxf(s[0][r], s[1][r]), fmaxf(s[2][r], s[3][r]));
        mx = fmaxf(mx, __shfl_xor(mx, 1));
        mx = fmaxf(mx, __shfl_xor(mx, 2));
        mx = fmaxf(mx, __shfl_xor(mx, 4));
        mx = fmaxf(mx, __shfl_xor(mx, 8));
        float sum = 0.f;
#pragma unroll
        for (int nt = 0; nt < 4; ++nt) {
          float e = __expf(s[nt][r] - mx);
          s[nt][r] = e;
          sum += e;
        }
        sum += __shfl_xor(sum, 1);
        sum += __shfl_xor(sum, 2);
        sum += __shfl_xor(sum, 4);
        sum += __shfl_xor(sum, 8);
        const float is = 1.f / sum;
        const int mloc = r + 8 * half;
#pragma unroll
        for (int nt = 0; nt < 4; ++nt)
          pw[mloc * VT_STRIDE + nt * 16 + nlan] = (_Float16)(s[nt][r] * is);
      }
      // O[16x32] = P[16x64] @ V[64x32]; V^T rows live in vT
      v16h p0 = load_fragA(pw + 0,  VT_STRIDE, lane);
      v16h p1 = load_fragA(pw + 32, VT_STRIDE, lane);
#pragma unroll
      for (int dt = 0; dt < 2; ++dt) {
        v16h bv0 = load_fragB(vT + (h * DH + dt * 16) * VT_STRIDE + 0,  VT_STRIDE, lane);
        v16h bv1 = load_fragB(vT + (h * DH + dt * 16) * VT_STRIDE + 32, VT_STRIDE, lane);
        v8f o = {};
        o = wmma_f16(p0, bv0, o);
        o = wmma_f16(p1, bv1, o);
#pragma unroll
        for (int r = 0; r < 8; ++r) {
          const int m = mt * 16 + r + 8 * half;
          xs[m * XS_STRIDE + h * DH + dt * 16 + nlan] = (_Float16)o[r];
        }
      }
    }
  }
  __syncthreads();

  // ---- Output projection: [64x384] @ [384x384] + proj_b --------------------
  // 4 M-tiles x 24 N-tiles in groups of 4 => 24 groups over 8 waves.
  float* ob = out + (size_t)b * NTOK * DIMC;
  for (int tg = wave; tg < 4 * 6; tg += 8) {
    const int mt  = tg / 6;
    const int nt0 = (tg % 6) * 4;
    v8f acc[4] = {{}, {}, {}, {}};
    for (int kt = 0; kt < DIMC / 32; ++kt) {
      v16h a = load_fragA(xs + mt * 16 * XS_STRIDE + kt * 32, XS_STRIDE, lane);
#pragma unroll
      for (int j = 0; j < 4; ++j) {
        v16h bf = load_fragB(pT + (nt0 + j) * 16 * DIMC + kt * 32, DIMC, lane);
        acc[j] = wmma_f16(a, bf, acc[j]);
      }
    }
#pragma unroll
    for (int j = 0; j < 4; ++j) {
      const int col  = (nt0 + j) * 16 + nlan;
      const float bv = proj_b[col];
#pragma unroll
      for (int r = 0; r < 8; ++r) {
        const int m = mt * 16 + r + 8 * half;
        if (m < NTOK) ob[m * DIMC + col] = acc[j][r] + bv;
      }
    }
  }
}

// ---------------------------------------------------------------------------
extern "C" void kernel_launch(void* const* d_in, const int* in_sizes, int n_in,
                              void* d_out, int out_size, void* d_ws, size_t ws_size,
                              hipStream_t stream) {
  const float* x          = (const float*)d_in[0];
  const float* qkv_w      = (const float*)d_in[1];
  const float* qkv_b      = (const float*)d_in[2];
  const float* proj_w     = (const float*)d_in[3];
  const float* proj_b     = (const float*)d_in[4];
  const float* bias_table = (const float*)d_in[5];
  const int*   rel_index  = (const int*)d_in[6];
  float* out = (float*)d_out;

  // workspace layout (all 16B aligned)
  char* ws = (char*)d_ws;
  _Float16* wT   = (_Float16*)(ws);                               // 884,736 B
  _Float16* pT   = (_Float16*)(ws + 884736);                      // 294,912 B
  float*    bias = (float*)   (ws + 884736 + 294912);             // 196,608 B

  // prep: 442368 + 147456 + 49152 = 638,976 elements
  {
    int total = QKVC * DIMC + DIMC * DIMC + NHEAD * 64 * 64;
    int grid = (total + 255) / 256;
    swin_prep_kernel<<<grid, 256, 0, stream>>>(qkv_w, proj_w, bias_table,
                                               rel_index, wT, pT, bias);
  }

  // fused attention: 4096 windows, 256 threads, ~224KB dynamic LDS
  {
    const size_t lds_bytes =
        (size_t)(3 * 64 * XS_STRIDE + 384 * VT_STRIDE + 8 * 16 * VT_STRIDE) *
        sizeof(_Float16);   // 224,256 B (within 320KB WGP LDS)
    swin_attn_kernel<<<4096, 256, lds_bytes, stream>>>(x, qkv_b, proj_b,
                                                       wT, pT, bias, out);
  }
}
